// IPMPEncoder_32487132627121
// MI455X (gfx1250) — matrix operand order repaired and checked by reading, not compiled, not run on hardware
//
#include <hip/hip_runtime.h>
#include <math.h>

#define B_ 2
#define L_ 2048
#define K_ 32
#define H_ 128
#define P_ 8
#define MSGK 480        // 456 padded to 15 k-tiles of 32
#define MSG_REAL 456
#define EPSC 1e-08f
#define LN_EPSC 1e-05f

typedef __attribute__((ext_vector_type(16))) __bf16 bf16x16;
typedef __attribute__((ext_vector_type(8)))  float  floatx8;

__device__ __forceinline__ unsigned short f2bf(float f) {
    unsigned int u = __float_as_uint(f);
    unsigned int r = u + 0x7FFFu + ((u >> 16) & 1u);
    return (unsigned short)(r >> 16);
}
__device__ __forceinline__ unsigned int pack2bf(float a, float b) {
    return (unsigned int)f2bf(a) | ((unsigned int)f2bf(b) << 16);
}
__device__ __forceinline__ float gelu_exact(float x) {
    return 0.5f * x * (1.0f + erff(x * 0.70710678118654752440f));
}

// ---- WMMA fragment loaders -------------------------------------------------
// A operand (16x32 bf16, MxK), per ISA 7.12.2: lane<16 holds K 0-7,16-23;
// lane>=16 holds K 8-15,24-31; row m = lane&15. Loaded from row-major LDS.
__device__ __forceinline__ bf16x16 load_a_frag(const unsigned short* lds,
                                               int stride, int row0, int kbase) {
    int lane = threadIdx.x & 31;
    int m = lane & 15, half = lane >> 4;
    const unsigned short* p = lds + (size_t)(row0 + m) * stride + kbase + half * 8;
    union { bf16x16 v; unsigned int u[8]; } r;
    const unsigned int* pa = (const unsigned int*)p;
    const unsigned int* pb = (const unsigned int*)(p + 16);
#pragma unroll
    for (int j = 0; j < 4; ++j) { r.u[j] = pa[j]; r.u[4 + j] = pb[j]; }
    return r.v;
}
// B operand: fragments are pre-swizzled in global scratch, 32 lanes x 16 bf16.
__device__ __forceinline__ bf16x16 load_b_frag(const unsigned short* gfrag) {
    int lane = threadIdx.x & 31;
    return ((const bf16x16*)gfrag)[lane];
}
__device__ __forceinline__ floatx8 wmma_bf16(bf16x16 a, bf16x16 b, floatx8 c) {
    return __builtin_amdgcn_wmma_f32_16x16x32_bf16(false, a, false, b, (short)0, c, false, false);
}

// ---- weight pre-swizzle: row-major fp32 KxN -> bf16 B-fragments ------------
// B layout (16-bit 32x16 KxN): lanes 0-15 hold K 0-7,16-23, lanes 16-31 hold K 8-15,24-31,
// column n = lane&15. Fragment index = ntile*KT + ktile, 512 bf16 per fragment.
__global__ void swizzle_kernel(const float* __restrict__ W, int K, int N, int KT,
                               unsigned short* __restrict__ dst) {
    int f = blockIdx.x;
    int ntile = f / KT, kt = f % KT;
    int lane = threadIdx.x;
    int n = ntile * 16 + (lane & 15);
    int half = lane >> 4;
    unsigned int* o = (unsigned int*)(dst + (size_t)f * 512 + lane * 16);
#pragma unroll
    for (int e = 0; e < 8; ++e) {
        int k0 = kt * 32 + half * 8 + (e >> 2) * 16 + 2 * (e & 3);
        float v0 = (k0 < K) ? W[(size_t)k0 * N + n] : 0.0f;
        float v1 = (k0 + 1 < K) ? W[(size_t)(k0 + 1) * N + n] : 0.0f;
        o[e] = pack2bf(v0, v1);
    }
}

// ---- frames ----------------------------------------------------------------
__global__ void frames_kernel(const float* __restrict__ X, float* __restrict__ Rb,
                              float* __restrict__ tb) {
    int i = blockIdx.x * blockDim.x + threadIdx.x;
    if (i >= B_ * L_) return;
    const float* x = X + (size_t)i * 12;
    float Nx = x[0], Ny = x[1], Nz = x[2];
    float Ax = x[3], Ay = x[4], Az = x[5];
    float Cx = x[6], Cy = x[7], Cz = x[8];
    float e0x = Ax - Nx, e0y = Ay - Ny, e0z = Az - Nz;
    float inv = rsqrtf(e0x * e0x + e0y * e0y + e0z * e0z + EPSC);
    e0x *= inv; e0y *= inv; e0z *= inv;
    float e1x = Cx - Ax, e1y = Cy - Ay, e1z = Cz - Az;
    float d = e0x * e1x + e0y * e1y + e0z * e1z;
    e1x -= e0x * d; e1y -= e0y * d; e1z -= e0z * d;
    inv = rsqrtf(e1x * e1x + e1y * e1y + e1z * e1z + EPSC);
    e1x *= inv; e1y *= inv; e1z *= inv;
    float e2x = e0y * e1z - e0z * e1y;
    float e2y = e0z * e1x - e0x * e1z;
    float e2z = e0x * e1y - e0y * e1x;
    float* R = Rb + (size_t)i * 9;
    R[0] = e0x; R[1] = e1x; R[2] = e2x;   // R[i][j], column j = e_j
    R[3] = e0y; R[4] = e1y; R[5] = e2y;
    R[6] = e0z; R[7] = e1z; R[8] = e2z;
    float* t = tb + (size_t)i * 3;
    t[0] = Ax; t[1] = Ay; t[2] = Az;
}

// ---- point projection: p_local = hV @ Wp + bp; p_global = R p_local + t ----
__global__ void proj_kernel(const float* __restrict__ hV, const float* __restrict__ Wp,
                            const float* __restrict__ bp, const float* __restrict__ Rb,
                            const float* __restrict__ tb, float* __restrict__ pl,
                            float* __restrict__ pg, float* __restrict__ plnorm) {
    int blk = blockIdx.x;
    int t = threadIdx.x;
    __shared__ alignas(16) float hv[H_];
    __shared__ float pls[24];
#pragma unroll
    for (int i = 0; i < 4; ++i) hv[t + 32 * i] = hV[(size_t)blk * H_ + t + 32 * i];
    __syncthreads();
    if (t < 24) {
        float s = bp[t];
        for (int c = 0; c < H_; ++c) s += hv[c] * Wp[c * 24 + t];
        pls[t] = s;
        pl[(size_t)blk * 24 + t] = s;
    }
    __syncthreads();
    if (t < P_) {
        float lx = pls[3 * t], ly = pls[3 * t + 1], lz = pls[3 * t + 2];
        plnorm[(size_t)blk * 8 + t] = sqrtf(lx * lx + ly * ly + lz * lz + EPSC);
        const float* R = Rb + (size_t)blk * 9;
        const float* tv = tb + (size_t)blk * 3;
        pg[(size_t)blk * 24 + 3 * t]     = R[0] * lx + R[1] * ly + R[2] * lz + tv[0];
        pg[(size_t)blk * 24 + 3 * t + 1] = R[3] * lx + R[4] * ly + R[5] * lz + tv[1];
        pg[(size_t)blk * 24 + 3 * t + 2] = R[6] * lx + R[7] * ly + R[8] * lz + tv[2];
    }
}

// ---- one 32xN GEMM step over the message MLP -------------------------------
__device__ __forceinline__ void gemm32(const unsigned short* A, int astride,
                                       const unsigned short* Bf, int KT, int wave,
                                       floatx8 (&acc)[2][2]) {
#pragma unroll
    for (int i = 0; i < 2; ++i)
#pragma unroll
        for (int j = 0; j < 2; ++j)
#pragma unroll
            for (int r = 0; r < 8; ++r) acc[i][j][r] = 0.0f;
    int n0 = wave * 2;
    for (int kt = 0; kt < KT; ++kt) {
        bf16x16 a0 = load_a_frag(A, astride, 0, kt * 32);
        bf16x16 a1 = load_a_frag(A, astride, 16, kt * 32);
        bf16x16 b0 = load_b_frag(Bf + ((size_t)(n0 * KT + kt)) * 512);
        bf16x16 b1 = load_b_frag(Bf + ((size_t)((n0 + 1) * KT + kt)) * 512);
        acc[0][0] = wmma_bf16(a0, b0, acc[0][0]);
        acc[0][1] = wmma_bf16(a0, b1, acc[0][1]);
        acc[1][0] = wmma_bf16(a1, b0, acc[1][0]);
        acc[1][1] = wmma_bf16(a1, b1, acc[1][1]);
    }
}

__device__ __forceinline__ void store_act(floatx8 (&acc)[2][2], const float* bias,
                                          unsigned short* dst, int wave) {
    int lane = threadIdx.x & 31;
    int n_in = lane & 15, half = lane >> 4;
#pragma unroll
    for (int mt = 0; mt < 2; ++mt)
#pragma unroll
        for (int nt = 0; nt < 2; ++nt) {
            int ncol = (wave * 2 + nt) * 16 + n_in;
            float bv = bias[ncol];
#pragma unroll
            for (int r = 0; r < 8; ++r) {
                int m = mt * 16 + half * 8 + r;
                dst[m * H_ + ncol] = f2bf(gelu_exact(acc[mt][nt][r] + bv));
            }
        }
}

// ---- fused message-MLP kernel (node mode / edge mode) ----------------------
__global__ __launch_bounds__(128) void msg_kernel(
    const float* __restrict__ hV, const float* __restrict__ hE,
    const int* __restrict__ Eidx, const float* __restrict__ Rb,
    const float* __restrict__ tb, const float* __restrict__ pl,
    const float* __restrict__ pg, const float* __restrict__ pln,
    const unsigned short* __restrict__ W1f, const float* __restrict__ b1,
    const unsigned short* __restrict__ W2f, const float* __restrict__ b2,
    const unsigned short* __restrict__ W3f, const float* __restrict__ b3,
    const float* __restrict__ mask_att, const float* __restrict__ lng,
    const float* __restrict__ lnb, float* __restrict__ outp, int edge_stage) {
    const int blk = blockIdx.x;          // b*L + l
    const int tid = threadIdx.x;
    const int wave = tid >> 5, lane = tid & 31;

    __shared__ alignas(16) unsigned char smem_pool[K_ * MSGK * 2]; // 30720 B
    unsigned short* mi = (unsigned short*)smem_pool;               // GEMM1 A
    float* out32 = (float*)smem_pool;                              // reused after GEMM1
    __shared__ alignas(16) unsigned short act1[K_ * H_];
    __shared__ alignas(16) unsigned short act2[K_ * H_];
    __shared__ alignas(16) float hv_row[H_];
    __shared__ float Rv[9], tv[3], plv[24], pgv[24], plnv[8];
    __shared__ float red[8];
    __shared__ float rpart[K_][8];

    hv_row[tid] = hV[(size_t)blk * H_ + tid];
    if (tid < 9) Rv[tid] = Rb[(size_t)blk * 9 + tid];
    if (tid < 3) tv[tid] = tb[(size_t)blk * 3 + tid];
    if (tid < 24) { plv[tid] = pl[(size_t)blk * 24 + tid]; pgv[tid] = pg[(size_t)blk * 24 + tid]; }
    if (tid < 8) plnv[tid] = pln[(size_t)blk * 8 + tid];
    __syncthreads();

    // ---- build message-input tile: 32 rows x 480 (bf16, zero padded) ------
    const int row = tid >> 2, sub = tid & 3;
    unsigned short* mrow = mi + row * MSGK;
    unsigned int* mrow32 = (unsigned int*)mrow;
    const int b = blk / L_;
    const int j = Eidx[(size_t)blk * K_ + row];
    const float* hErow = hE + ((size_t)blk * K_ + row) * H_;
    const float* nbhV = hV + ((size_t)(b * L_ + j)) * H_;
    const float4* hE4 = (const float4*)hErow;
    const float4* nb4 = (const float4*)nbhV;
    const float4* hv4 = (const float4*)hv_row;
#pragma unroll
    for (int i = 0; i < 8; ++i) {
        int c4 = sub * 8 + i;                 // float4 index; element c = 4*c4
        float4 a = hv4[c4];
        float4 e = hE4[c4];
        float4 nv = nb4[c4];
        mrow32[2 * c4]            = pack2bf(a.x, a.y);
        mrow32[2 * c4 + 1]        = pack2bf(a.z, a.w);
        mrow32[64 + 2 * c4]       = pack2bf(e.x, e.y);     // cols 128..255
        mrow32[64 + 2 * c4 + 1]   = pack2bf(e.z, e.w);
        mrow32[128 + 2 * c4]      = pack2bf(nv.x, nv.y);   // cols 256..383
        mrow32[128 + 2 * c4 + 1]  = pack2bf(nv.z, nv.w);
    }
    if (sub == 0) {
#pragma unroll
        for (int i = 0; i < 12; ++i) mrow32[192 + i] = pack2bf(plv[2 * i], plv[2 * i + 1]);
#pragma unroll
        for (int i = 0; i < 4; ++i)  mrow32[204 + i] = pack2bf(plnv[2 * i], plnv[2 * i + 1]);
#pragma unroll
        for (int i = 228; i < 240; ++i) mrow32[i] = 0u;    // cols 456..479
    } else if (sub == 1) {
        const float* npg = pg + ((size_t)(b * L_ + j)) * 24;
        float nl[24], nn[8], gg[8];
#pragma unroll
        for (int p = 0; p < P_; ++p) {
            float dx = npg[3 * p] - tv[0], dy = npg[3 * p + 1] - tv[1], dz = npg[3 * p + 2] - tv[2];
            float lx = Rv[0] * dx + Rv[3] * dy + Rv[6] * dz;   // R^T d
            float ly = Rv[1] * dx + Rv[4] * dy + Rv[7] * dz;
            float lz = Rv[2] * dx + Rv[5] * dy + Rv[8] * dz;
            nl[3 * p] = lx; nl[3 * p + 1] = ly; nl[3 * p + 2] = lz;
            nn[p] = sqrtf(lx * lx + ly * ly + lz * lz + EPSC);
            float gx = pgv[3 * p] - npg[3 * p];
            float gy = pgv[3 * p + 1] - npg[3 * p + 1];
            float gz = pgv[3 * p + 2] - npg[3 * p + 2];
            gg[p] = sqrtf(gx * gx + gy * gy + gz * gz + EPSC);
        }
#pragma unroll
        for (int i = 0; i < 12; ++i) mrow32[208 + i] = pack2bf(nl[2 * i], nl[2 * i + 1]); // 416..439
#pragma unroll
        for (int i = 0; i < 4; ++i)  mrow32[220 + i] = pack2bf(nn[2 * i], nn[2 * i + 1]); // 440..447
#pragma unroll
        for (int i = 0; i < 4; ++i)  mrow32[224 + i] = pack2bf(gg[2 * i], gg[2 * i + 1]); // 448..455
    }
    __syncthreads();

    // ---- 3-layer MLP on the 32x456 tile via WMMA ---------------------------
    floatx8 acc[2][2];
    gemm32(mi, MSGK, W1f, 15, wave, acc);      // (32x480)@(480x128)
    store_act(acc, b1, act1, wave);
    __syncthreads();
    gemm32(act1, H_, W2f, 4, wave, acc);       // (32x128)@(128x128)
    store_act(acc, b2, act2, wave);
    __syncthreads();
    gemm32(act2, H_, W3f, 4, wave, acc);       // (32x128)@(128x128)
    {
        int n_in = lane & 15, half = lane >> 4;
#pragma unroll
        for (int mt = 0; mt < 2; ++mt)
#pragma unroll
            for (int nt = 0; nt < 2; ++nt) {
                int ncol = (wave * 2 + nt) * 16 + n_in;
                float bv = b3[ncol];
#pragma unroll
                for (int r = 0; r < 8; ++r) {
                    int m = mt * 16 + half * 8 + r;
                    float x = acc[mt][nt][r] + bv;
                    if (!edge_stage) x *= mask_att[(size_t)blk * K_ + m];
                    out32[m * H_ + ncol] = x;
                }
            }
    }
    __syncthreads();

    if (!edge_stage) {
        // mean over K, residual + LayerNorm -> h1
        float s = 0.0f;
        for (int k2 = 0; k2 < K_; ++k2) s += out32[k2 * H_ + tid];
        float h = hv_row[tid] + s * (1.0f / (float)K_);
        float v1 = h, v2 = h * h;
        for (int off = 16; off > 0; off >>= 1) {
            v1 += __shfl_xor(v1, off, 32);
            v2 += __shfl_xor(v2, off, 32);
        }
        if (lane == 0) { red[wave] = v1; red[4 + wave] = v2; }
        __syncthreads();
        float S = red[0] + red[1] + red[2] + red[3];
        float SS = red[4] + red[5] + red[6] + red[7];
        float mu = S * (1.0f / (float)H_);
        float rstd = rsqrtf(SS * (1.0f / (float)H_) - mu * mu + LN_EPSC);
        outp[(size_t)blk * H_ + tid] = (h - mu) * rstd * lng[tid] + lnb[tid];
    } else {
        // per-row LN(h_E + edge_m) -> h_E out (vectorized)
        const float4* o4 = (const float4*)(out32 + row * H_);
        float ps = 0.0f, pss = 0.0f;
#pragma unroll
        for (int i = 0; i < 8; ++i) {
            int c4 = sub * 8 + i;
            float4 e = hE4[c4];
            float4 m = o4[c4];
            float x0 = e.x + m.x, x1 = e.y + m.y, x2 = e.z + m.z, x3 = e.w + m.w;
            ps += x0 + x1 + x2 + x3;
            pss += x0 * x0 + x1 * x1 + x2 * x2 + x3 * x3;
        }
        rpart[row][sub] = ps; rpart[row][4 + sub] = pss;
        __syncthreads();
        float S = rpart[row][0] + rpart[row][1] + rpart[row][2] + rpart[row][3];
        float SS = rpart[row][4] + rpart[row][5] + rpart[row][6] + rpart[row][7];
        float mu = S * (1.0f / (float)H_);
        float rstd = rsqrtf(SS * (1.0f / (float)H_) - mu * mu + LN_EPSC);
        float4* oE4 = (float4*)(outp + ((size_t)blk * K_ + row) * H_);
#pragma unroll
        for (int i = 0; i < 8; ++i) {
            int c4 = sub * 8 + i;
            int c = 4 * c4;
            float4 e = hE4[c4];
            float4 m = o4[c4];
            float4 y;
            y.x = (e.x + m.x - mu) * rstd * lng[c]     + lnb[c];
            y.y = (e.y + m.y - mu) * rstd * lng[c + 1] + lnb[c + 1];
            y.z = (e.z + m.z - mu) * rstd * lng[c + 2] + lnb[c + 2];
            y.w = (e.w + m.w - mu) * rstd * lng[c + 3] + lnb[c + 3];
            oE4[c4] = y;
        }
    }
}

// ---- FFN: h1 -> gelu(h1@Wd_in) @ Wd_out, residual, LN2, mask_V -------------
__global__ __launch_bounds__(256) void ffn_kernel(
    const float* __restrict__ h1, const unsigned short* __restrict__ Winf,
    const float* __restrict__ bdin, const unsigned short* __restrict__ Woutf,
    const float* __restrict__ bdout, const float* __restrict__ g2,
    const float* __restrict__ be2, const float* __restrict__ maskV,
    float* __restrict__ hv_out) {
    const int tid = threadIdx.x, wave = tid >> 5, lane = tid & 31;
    const int row0 = blockIdx.x * 32;
    __shared__ alignas(16) unsigned short h1bf[32 * H_];
    __shared__ alignas(16) float h1f[32 * H_];
    __shared__ alignas(16) unsigned short hid[32 * 512];
    __shared__ float mu_s[32], rstd_s[32];
    float* out32 = (float*)hid;   // reused after GEMM-B reads complete

    {
        const float4* src = (const float4*)(h1 + (size_t)row0 * H_);
        float4* dstf = (float4*)h1f;
        unsigned int* dstb = (unsigned int*)h1bf;
#pragma unroll
        for (int i = 0; i < 4; ++i) {
            int idx4 = tid + i * 256;              // float4 index
            float4 v = src[idx4];
            dstf[idx4] = v;
            dstb[2 * idx4]     = pack2bf(v.x, v.y);
            dstb[2 * idx4 + 1] = pack2bf(v.z, v.w);
        }
    }
    __syncthreads();

    // GEMM A: (32x128)@(128x512); each wave owns 64 output cols
    floatx8 acc[2][4];
#pragma unroll
    for (int i = 0; i < 2; ++i)
#pragma unroll
        for (int j = 0; j < 4; ++j)
#pragma unroll
            for (int r = 0; r < 8; ++r) acc[i][j][r] = 0.0f;
    for (int kt = 0; kt < 4; ++kt) {
        bf16x16 a0 = load_a_frag(h1bf, H_, 0, kt * 32);
        bf16x16 a1 = load_a_frag(h1bf, H_, 16, kt * 32);
#pragma unroll
        for (int nt = 0; nt < 4; ++nt) {
            int n = wave * 4 + nt;
            bf16x16 bf = load_b_frag(Winf + ((size_t)(n * 4 + kt)) * 512);
            acc[0][nt] = wmma_bf16(a0, bf, acc[0][nt]);
            acc[1][nt] = wmma_bf16(a1, bf, acc[1][nt]);
        }
    }
    {
        int n_in = lane & 15, half = lane >> 4;
#pragma unroll
        for (int mt = 0; mt < 2; ++mt)
#pragma unroll
            for (int nt = 0; nt < 4; ++nt) {
                int ncol = (wave * 4 + nt) * 16 + n_in;
                float bv = bdin[ncol];
#pragma unroll
                for (int r = 0; r < 8; ++r) {
                    int m = mt * 16 + half * 8 + r;
                    hid[m * 512 + ncol] = f2bf(gelu_exact(acc[mt][nt][r] + bv));
                }
            }
    }
    __syncthreads();

    // GEMM B: (32x512)@(512x128); each wave owns 16 output cols
    floatx8 c0, c1;
#pragma unroll
    for (int r = 0; r < 8; ++r) { c0[r] = 0.0f; c1[r] = 0.0f; }
    for (int kt = 0; kt < 16; ++kt) {
        bf16x16 a0 = load_a_frag(hid, 512, 0, kt * 32);
        bf16x16 a1 = load_a_frag(hid, 512, 16, kt * 32);
        bf16x16 bf = load_b_frag(Woutf + ((size_t)(wave * 16 + kt)) * 512);
        c0 = wmma_bf16(a0, bf, c0);
        c1 = wmma_bf16(a1, bf, c1);
    }
    __syncthreads();   // all hid reads done; reuse as out32
    {
        int n_in = lane & 15, half = lane >> 4;
        int ncol = wave * 16 + n_in;
        float bv = bdout[ncol];
#pragma unroll
        for (int r = 0; r < 8; ++r) {
            int m = half * 8 + r;
            out32[m * H_ + ncol] = c0[r] + bv + h1f[m * H_ + ncol];
            int m2 = 16 + half * 8 + r;
            out32[m2 * H_ + ncol] = c1[r] + bv + h1f[m2 * H_ + ncol];
        }
    }
    __syncthreads();
    if (tid < 32) {
        float S = 0.0f, SS = 0.0f;
        const float* rw = out32 + tid * H_;
        for (int c = 0; c < H_; ++c) { float x = rw[c]; S += x; SS += x * x; }
        float mu = S * (1.0f / (float)H_);
        mu_s[tid] = mu;
        rstd_s[tid] = rsqrtf(SS * (1.0f / (float)H_) - mu * mu + LN_EPSC);
    }
    __syncthreads();
    {
        float4* dst = (float4*)(hv_out + (size_t)row0 * H_);
        const float4* src = (const float4*)out32;
#pragma unroll
        for (int i = 0; i < 4; ++i) {
            int idx4 = tid + i * 256;              // float4 index
            int r = idx4 >> 5, c = (idx4 & 31) * 4;
            int grow = row0 + r;
            float mv = maskV[grow];
            float4 v = src[idx4];
            float4 y;
            y.x = ((v.x - mu_s[r]) * rstd_s[r] * g2[c]     + be2[c])     * mv;
            y.y = ((v.y - mu_s[r]) * rstd_s[r] * g2[c + 1] + be2[c + 1]) * mv;
            y.z = ((v.z - mu_s[r]) * rstd_s[r] * g2[c + 2] + be2[c + 2]) * mv;
            y.w = ((v.w - mu_s[r]) * rstd_s[r] * g2[c + 3] + be2[c + 3]) * mv;
            dst[idx4] = y;
        }
    }
}

// ---------------------------------------------------------------------------
extern "C" void kernel_launch(void* const* d_in, const int* in_sizes, int n_in,
                              void* d_out, int out_size, void* d_ws, size_t ws_size,
                              hipStream_t stream) {
    (void)in_sizes; (void)n_in; (void)out_size; (void)ws_size;
    const float* h_V   = (const float*)d_in[0];
    const float* h_E   = (const float*)d_in[1];
    const int*   E_idx = (const int*)d_in[2];
    const float* X     = (const float*)d_in[3];
    const float* maskV = (const float*)d_in[4];
    const float* maskA = (const float*)d_in[5];
    const float* Wp_n  = (const float*)d_in[6];
    const float* bp_n  = (const float*)d_in[7];
    const float* Wp_e  = (const float*)d_in[8];
    const float* bp_e  = (const float*)d_in[9];
    const float* W1  = (const float*)d_in[10]; const float* b1  = (const float*)d_in[11];
    const float* W2  = (const float*)d_in[12]; const float* b2  = (const float*)d_in[13];
    const float* W3  = (const float*)d_in[14]; const float* b3  = (const float*)d_in[15];
    const float* W11 = (const float*)d_in[16]; const float* b11 = (const float*)d_in[17];
    const float* W12 = (const float*)d_in[18]; const float* b12 = (const float*)d_in[19];
    const float* W13 = (const float*)d_in[20]; const float* b13 = (const float*)d_in[21];
    const float* Wdi = (const float*)d_in[22]; const float* bdi = (const float*)d_in[23];
    const float* Wdo = (const float*)d_in[24]; const float* bdo = (const float*)d_in[25];
    const float* g1  = (const float*)d_in[26]; const float* be1 = (const float*)d_in[27];
    const float* g2  = (const float*)d_in[28]; const float* be2 = (const float*)d_in[29];
    const float* g3  = (const float*)d_in[30]; const float* be3 = (const float*)d_in[31];

    const int BL = B_ * L_;
    char* w = (char*)d_ws;
    auto alloc = [&](size_t bytes) -> void* {
        void* p = (void*)w;
        w += (bytes + 255) & ~(size_t)255;
        return p;
    };
    float* Rb   = (float*)alloc((size_t)BL * 9 * 4);
    float* tb   = (float*)alloc((size_t)BL * 3 * 4);
    float* pl_n = (float*)alloc((size_t)BL * 24 * 4);
    float* pg_n = (float*)alloc((size_t)BL * 24 * 4);
    float* pn_n = (float*)alloc((size_t)BL * 8 * 4);
    float* pl_e = (float*)alloc((size_t)BL * 24 * 4);
    float* pg_e = (float*)alloc((size_t)BL * 24 * 4);
    float* pn_e = (float*)alloc((size_t)BL * 8 * 4);
    float* h1   = (float*)alloc((size_t)BL * H_ * 4);
    unsigned short* W1f  = (unsigned short*)alloc((size_t)15 * 8 * 512 * 2);
    unsigned short* W2f  = (unsigned short*)alloc((size_t)4 * 8 * 512 * 2);
    unsigned short* W3f  = (unsigned short*)alloc((size_t)4 * 8 * 512 * 2);
    unsigned short* W11f = (unsigned short*)alloc((size_t)15 * 8 * 512 * 2);
    unsigned short* W12f = (unsigned short*)alloc((size_t)4 * 8 * 512 * 2);
    unsigned short* W13f = (unsigned short*)alloc((size_t)4 * 8 * 512 * 2);
    unsigned short* Wif  = (unsigned short*)alloc((size_t)4 * 32 * 512 * 2);
    unsigned short* Wof  = (unsigned short*)alloc((size_t)16 * 8 * 512 * 2);

    float* hv_out = (float*)d_out;
    float* he_out = hv_out + (size_t)BL * H_;

    // weight pre-swizzle to WMMA B-fragment layout (bf16)
    swizzle_kernel<<<15 * 8, 32, 0, stream>>>(W1, 456, 128, 15, W1f);
    swizzle_kernel<<<4 * 8, 32, 0, stream>>>(W2, 128, 128, 4, W2f);
    swizzle_kernel<<<4 * 8, 32, 0, stream>>>(W3, 128, 128, 4, W3f);
    swizzle_kernel<<<15 * 8, 32, 0, stream>>>(W11, 456, 128, 15, W11f);
    swizzle_kernel<<<4 * 8, 32, 0, stream>>>(W12, 128, 128, 4, W12f);
    swizzle_kernel<<<4 * 8, 32, 0, stream>>>(W13, 128, 128, 4, W13f);
    swizzle_kernel<<<4 * 32, 32, 0, stream>>>(Wdi, 128, 512, 4, Wif);
    swizzle_kernel<<<16 * 8, 32, 0, stream>>>(Wdo, 512, 128, 16, Wof);

    frames_kernel<<<(BL + 255) / 256, 256, 0, stream>>>(X, Rb, tb);
    proj_kernel<<<BL, 32, 0, stream>>>(h_V, Wp_n, bp_n, Rb, tb, pl_n, pg_n, pn_n);
    msg_kernel<<<BL, 128, 0, stream>>>(h_V, h_E, E_idx, Rb, tb, pl_n, pg_n, pn_n,
                                       W1f, b1, W2f, b2, W3f, b3, maskA, g1, be1,
                                       h1, 0);
    ffn_kernel<<<BL / 32, 256, 0, stream>>>(h1, Wif, bdi, Wof, bdo, g2, be2, maskV,
                                            hv_out);
    proj_kernel<<<BL, 32, 0, stream>>>(hv_out, Wp_e, bp_e, Rb, tb, pl_e, pg_e, pn_e);
    msg_kernel<<<BL, 128, 0, stream>>>(hv_out, h_E, E_idx, Rb, tb, pl_e, pg_e, pn_e,
                                       W11f, b11, W12f, b12, W13f, b13, nullptr,
                                       g3, be3, he_out, 1);
}
